// fast_BEV_10488310137171
// MI455X (gfx1250) — matrix operand order repaired and checked by reading, not compiled, not run on hardware
//
#include <hip/hip_runtime.h>
#include <hip/hip_bf16.h>

// ---------------- constants ----------------
#define BATCH   2
#define NCAM    6
#define IMG_H   256
#define IMG_W   704
#define FH      32
#define FW_     88
#define NPIX    (FH*FW_)          // 2816
#define CFEAT   256
#define NPTS    120000
#define NIMG    (BATCH*NCAM)
#define C1H     64
#define C1W     176
#define EPSBN   1e-5f
#define KFUSE   320
#define SA_STRIDE 264             // 256 + 8 bf16 pad -> 528B rows, bank-conflict free
#define SW_STRIDE 264

typedef __bf16        v16bf __attribute__((ext_vector_type(16)));
typedef float         v8f   __attribute__((ext_vector_type(8)));
typedef float         v4f   __attribute__((ext_vector_type(4)));
typedef unsigned int  v4u   __attribute__((ext_vector_type(4)));
typedef unsigned int  v8u   __attribute__((ext_vector_type(8)));
typedef int           v8i   __attribute__((ext_vector_type(8)));
typedef int           v4i   __attribute__((ext_vector_type(4)));

static __device__ inline __bf16 f2bf(float f) {
    unsigned u = __builtin_bit_cast(unsigned, f);
    u += 0x7FFFu + ((u >> 16) & 1u);              // round-to-nearest-even
    unsigned short h = (unsigned short)(u >> 16);
    return __builtin_bit_cast(__bf16, h);
}

// load a 16-bf16 WMMA fragment as two 16B chunks (elems 0..7 from p0, 8..15 from p1)
static __device__ inline v16bf ld_frag(const __bf16* p0, const __bf16* p1) {
    v4u lo = *(const v4u*)p0;
    v4u hi = *(const v4u*)p1;
    v8u c  = __builtin_shufflevector(lo, hi, 0, 1, 2, 3, 4, 5, 6, 7);
    return __builtin_bit_cast(v16bf, c);
}
static __device__ inline v16bf zero_frag() {
    v8u z = {0, 0, 0, 0, 0, 0, 0, 0};
    return __builtin_bit_cast(v16bf, z);
}

// ---------------- 0: fold aug/calib into one 3x4 projection per (b,cam) ----------------
__global__ void precompute_proj(const float* __restrict__ l2i,
                                const float* __restrict__ img_aug,
                                const float* __restrict__ lidar_aug,
                                float* __restrict__ proj_out) {
    int id = threadIdx.x;
    if (id >= BATCH * NCAM) return;
    int b = id / NCAM;
    const float* L = l2i       + (size_t)id * 16;
    const float* A = img_aug   + (size_t)id * 16;
    const float* G = lidar_aug + (size_t)b * 16;
    float Ar[4][4];
    for (int r = 0; r < 4; ++r)
        for (int c = 0; c < 4; ++c) Ar[r][c] = A[r*4+c];
    Ar[0][3] = 0.f; Ar[1][3] = 0.f; Ar[2][3] = 0.f;   // img_aug with translation zeroed
    float P[3][4];
    for (int r = 0; r < 3; ++r)
        for (int c = 0; c < 4; ++c) {
            float s = 0.f;
            for (int k = 0; k < 4; ++k) s += Ar[r][k] * L[k*4+c];
            P[r][c] = s;
        }
    float R[3][3], t[3];
    for (int r = 0; r < 3; ++r) { for (int c = 0; c < 3; ++c) R[r][c] = G[r*4+c]; t[r] = G[r*4+3]; }
    float* O = proj_out + (size_t)id * 16;
    for (int r = 0; r < 3; ++r) {
        float m0 = 0.f, m1 = 0.f, m2 = 0.f;
        for (int k = 0; k < 3; ++k) { m0 += P[r][k]*R[0][k]; m1 += P[r][k]*R[1][k]; m2 += P[r][k]*R[2][k]; }
        float c3 = P[r][3] - (m0*t[0] + m1*t[1] + m2*t[2]);
        O[r*4+0] = m0; O[r*4+1] = m1; O[r*4+2] = m2; O[r*4+3] = c3;
    }
    O[12] = A[3];       // iat_u
    O[13] = A[7];       // iat_v
}

// ---------------- 1: scatter lidar points into depth images ----------------
__global__ void scatter_depth(const float* __restrict__ pts,
                              const float* __restrict__ proj,
                              float* __restrict__ depth) {
    int gid = blockIdx.x * blockDim.x + threadIdx.x;
    const int total = BATCH * NPTS * NCAM;
    if (gid >= total) return;
    int cam = gid % NCAM;
    int pi  = (gid / NCAM) % NPTS;
    int b   = gid / (NCAM * NPTS);
    const float* p = pts + ((size_t)b * NPTS + pi) * 4;
    float x = p[0], y = p[1], z = p[2];
    const float* M = proj + (size_t)(b * NCAM + cam) * 16;
    float q0 = M[0]*x + M[1]*y + M[2]*z  + M[3];
    float q1 = M[4]*x + M[5]*y + M[6]*z  + M[7];
    float q2 = M[8]*x + M[9]*y + M[10]*z + M[11];
    float u = q0 / q2 + M[12];
    float v = q1 / q2 + M[13];
    if (u >= 0.f && v >= 0.f && u < (float)IMG_W && v < (float)IMG_H && q2 > 0.f) {
        int ui = (int)u, vi = (int)v;
        depth[((size_t)(b * NCAM + cam) * IMG_H + vi) * IMG_W + ui] = q2;
    }
}

// ---------------- 2: conv1 5x5 s4 p2, 1->32, + BN + ReLU ----------------
__global__ void conv1_kernel(const float* __restrict__ depth, const float* __restrict__ w,
                             const float* __restrict__ bias, const float* __restrict__ g,
                             const float* __restrict__ be, const float* __restrict__ mn,
                             const float* __restrict__ vr, float* __restrict__ out) {
    int gid = blockIdx.x * blockDim.x + threadIdx.x;
    const int total = NIMG * 32 * C1H * C1W;
    if (gid >= total) return;
    int ox  = gid % C1W;
    int oy  = (gid / C1W) % C1H;
    int oc  = (gid / (C1W * C1H)) % 32;
    int img = gid / (C1W * C1H * 32);
    const float* dp = depth + (size_t)img * IMG_H * IMG_W;
    const float* wp = w + oc * 25;
    float acc = 0.f;
    for (int ky = 0; ky < 5; ++ky) {
        int iy = oy * 4 - 2 + ky;
        if (iy < 0 || iy >= IMG_H) continue;
        for (int kx = 0; kx < 5; ++kx) {
            int ix = ox * 4 - 2 + kx;
            if (ix < 0 || ix >= IMG_W) continue;
            acc += dp[iy * IMG_W + ix] * wp[ky * 5 + kx];
        }
    }
    float inv = rsqrtf(vr[oc] + EPSBN);
    float r = ((acc + bias[oc]) - mn[oc]) * g[oc] * inv + be[oc];
    out[gid] = fmaxf(r, 0.f);
}

// ---------------- 3: conv2 5x5 s2 p2, 32->64, + BN + ReLU ----------------
__global__ void conv2_kernel(const float* __restrict__ d1, const float* __restrict__ w,
                             const float* __restrict__ bias, const float* __restrict__ g,
                             const float* __restrict__ be, const float* __restrict__ mn,
                             const float* __restrict__ vr, float* __restrict__ out) {
    int gid = blockIdx.x * blockDim.x + threadIdx.x;
    const int total = NIMG * 64 * FH * FW_;
    if (gid >= total) return;
    int ox  = gid % FW_;
    int oy  = (gid / FW_) % FH;
    int oc  = (gid / (FW_ * FH)) % 64;
    int img = gid / (FW_ * FH * 64);
    const float* ip = d1 + (size_t)img * 32 * C1H * C1W;
    float acc = 0.f;
    for (int ci = 0; ci < 32; ++ci) {
        const float* wp  = w + ((size_t)oc * 32 + ci) * 25;
        const float* ipc = ip + (size_t)ci * C1H * C1W;
        for (int ky = 0; ky < 5; ++ky) {
            int iy = oy * 2 - 2 + ky;
            if (iy < 0 || iy >= C1H) continue;
            for (int kx = 0; kx < 5; ++kx) {
                int ix = ox * 2 - 2 + kx;
                if (ix < 0 || ix >= C1W) continue;
                acc += ipc[iy * C1W + ix] * wp[ky * 5 + kx];
            }
        }
    }
    float inv = rsqrtf(vr[oc] + EPSBN);
    float r = ((acc + bias[oc]) - mn[oc]) * g[oc] * inv + be[oc];
    out[gid] = fmaxf(r, 0.f);
}

// ---------------- 4a: pixel-major bf16 activation staging: act[img][pixel][320] ----------------
__global__ void build_act(const float* __restrict__ d2, const float* __restrict__ feat,
                          __bf16* __restrict__ act) {
    int gid = blockIdx.x * blockDim.x + threadIdx.x;
    const int total = NIMG * NPIX * KFUSE;
    if (gid >= total) return;
    int c   = gid % KFUSE;
    int p   = (gid / KFUSE) % NPIX;
    int img = gid / (KFUSE * NPIX);
    float v = (c < 64) ? d2[((size_t)img * 64 + c) * NPIX + p]
                       : feat[((size_t)img * CFEAT + (c - 64)) * NPIX + p];
    act[gid] = f2bf(v);
}

// ---------------- 4b: weight staging: wT[tap][o][k] bf16 and wbf[o][k] bf16 ----------------
__global__ void convert_weights(const float* __restrict__ fw, const float* __restrict__ finw,
                                __bf16* __restrict__ wT, __bf16* __restrict__ wbf) {
    int gid = blockIdx.x * blockDim.x + threadIdx.x;
    const int NWT = 9 * 256 * KFUSE;
    const int NWF = 80 * 256;
    if (gid < NWT) {
        int k   = gid % KFUSE;
        int o   = (gid / KFUSE) % 256;
        int tap = gid / (KFUSE * 256);
        wT[gid] = f2bf(fw[((size_t)o * KFUSE + k) * 9 + tap]);
    } else if (gid < NWT + NWF) {
        int i = gid - NWT;
        wbf[i] = f2bf(finw[i]);
    }
}

// ---------------- 5: fuse conv 3x3 s1 p1, 320->256, + BN  (WMMA implicit GEMM) ----------------
// grid (176 Mtiles, 16 Ntiles, 12 images), block = 1 wave
__global__ void __launch_bounds__(32)
fuse_conv_wmma(const __bf16* __restrict__ act, const __bf16* __restrict__ wT,
               const float* __restrict__ fb, const float* __restrict__ fg,
               const float* __restrict__ fbe, const float* __restrict__ fm,
               const float* __restrict__ fv, float* __restrict__ fuse) {
    int mtile = blockIdx.x;
    int ntile = blockIdx.y;
    int img   = blockIdx.z;
    int lane  = threadIdx.x;
    int mrow = mtile * 16 + (lane & 15);
    int py = mrow / FW_, px = mrow % FW_;
    int ocol   = ntile * 16 + (lane & 15);
    int khalfA = (lane < 16) ? 0 : 8;     // ISA 16-bit A layout K offset
    int kbaseB = (lane < 16) ? 0 : 16;    // unused; B loads with khalfA, same k order as A
    const __bf16* actimg = act + (size_t)img * NPIX * KFUSE;
    v8f acc = {};
    for (int tap = 0; tap < 9; ++tap) {
        int dy = tap / 3 - 1, dx = tap % 3 - 1;
        int sy = py + dy, sx = px + dx;
        bool inb = (sy >= 0 && sy < FH && sx >= 0 && sx < FW_);
        const __bf16* arow = actimg + (size_t)(sy * FW_ + sx) * KFUSE;
        const __bf16* brow = wT + ((size_t)tap * 256 + ocol) * KFUSE;
        for (int K0 = 0; K0 < KFUSE; K0 += 32) {
            v16bf a = inb ? ld_frag(arow + K0 + khalfA, arow + K0 + khalfA + 16)
                          : zero_frag();
            v16bf bm = ld_frag(brow + K0 + khalfA, brow + K0 + khalfA + 16);
            acc = __builtin_amdgcn_wmma_f32_16x16x32_bf16(false, a, false, bm,
                                                          (short)0, acc, false, false);
        }
    }
    int o = ocol;
    float inv   = rsqrtf(fv[o] + EPSBN);
    float scale = fg[o] * inv;
    float shift = fbe[o] + (fb[o] - fm[o]) * scale;
    int mbase = mtile * 16 + ((lane < 16) ? 0 : 8);
    float* outp = fuse + ((size_t)img * CFEAT + o) * NPIX + mbase;
    v4f o0 = {acc[0]*scale+shift, acc[1]*scale+shift, acc[2]*scale+shift, acc[3]*scale+shift};
    v4f o1 = {acc[4]*scale+shift, acc[5]*scale+shift, acc[6]*scale+shift, acc[7]*scale+shift};
    *(v4f*)outp       = o0;
    *(v4f*)(outp + 4) = o1;
}

// ---------------- 6: fused voxel gather + z-sum + final 1x1 conv (WMMA) + BN/ReLU + attention ----
// TDM preloads the 80x256 bf16 weight matrix into LDS with per-512B-row +16B padding.
__global__ void __launch_bounds__(256)
bev_final_kernel(const float* __restrict__ fuse, const float* __restrict__ proj,
                 const __bf16* __restrict__ wbf,
                 const float* __restrict__ bfin, const float* __restrict__ fng,
                 const float* __restrict__ fnb, const float* __restrict__ fnm,
                 const float* __restrict__ fnv, const float* __restrict__ attw,
                 const float* __restrict__ attb, float* __restrict__ out) {
    __shared__ int    s_cam[16][4];
    __shared__ int    s_off[16][4];
    __shared__ __bf16 s_a[16 * SA_STRIDE];   // gathered vol tile (padded rows)
    __shared__ __bf16 s_w[80 * SW_STRIDE];   // TDM-loaded final weights (padded rows)
    __shared__ float  s_x[16 * 80];
    __shared__ float  s_att[16];
    int bid  = blockIdx.x;
    int b    = bid / 2500;
    int tile = bid % 2500;
    int tid  = threadIdx.x;
    int wid  = tid >> 5;
    int lane = tid & 31;

    // --- issue TDM from wave 0: 80*256 bf16 (40960B) -> LDS, pad 4 dwords per 128 dwords
    if (wid == 0) {
        unsigned lds_off = (unsigned)(size_t)(void*)&s_w[0];
        unsigned long long ga = (unsigned long long)(size_t)(const void*)wbf;
        v4u g0 = { 1u,                                   // count=1 (valid descriptor)
                   lds_off,                              // lds_addr
                   (unsigned)(ga & 0xFFFFFFFFu),         // global_addr lo
                   (unsigned)((ga >> 32) & 0x1FFFFFFu) | (2u << 30) }; // addr hi | type=2
        v8i g1 = { (int)0x07910000,   // data_size=2B, pad_enable, interval=128dw, amount=4dw
                   (int)0x50000000,   // tensor_dim0 = 20480 (low16 in [31:16])
                   (int)0x00010000,   // tensor_dim1 = 1
                   (int)0x50000000,   // tile_dim0 = 20480
                   1,                 // tile_dim1 = 1
                   20480,             // tensor_dim0_stride
                   0, 0 };
        v4i zz = {0, 0, 0, 0};
        v8i z8 = {0, 0, 0, 0, 0, 0, 0, 0};
        __builtin_amdgcn_tensor_load_to_lds(g0, g1, zz, zz, z8, 0);
    }

    // --- phase 1: voxel projection for 16 pixels x 4 z-slices
    if (tid < 64) {
        int m = tid >> 2, iz = tid & 3;
        int p  = tile * 16 + m;
        int ix = p / 200, iy = p % 200;
        float wx = ix * 0.5f - 50.0f;
        float wy = iy * 0.5f - 50.0f;
        float wz = iz * 1.5f - 4.7f;
        int sel = -1, off = 0;
        for (int cam = 0; cam < NCAM; ++cam) {     // last valid camera wins
            const float* M = proj + (size_t)(b * NCAM + cam) * 16;
            float q0 = M[0]*wx + M[1]*wy + M[2]*wz  + M[3];
            float q1 = M[4]*wx + M[5]*wy + M[6]*wz  + M[7];
            float q2 = M[8]*wx + M[9]*wy + M[10]*wz + M[11];
            float u = q0 / q2 + M[12];
            float v = q1 / q2 + M[13];
            int ufm = (int)rintf(u * 0.125f);
            int vfm = (int)rintf(v * 0.125f);
            if (ufm >= 0 && vfm >= 0 && ufm < FW_ && vfm < FH && q2 > 0.f) {
                sel = cam; off = vfm * FW_ + ufm;
            }
        }
        s_cam[m][iz] = sel; s_off[m][iz] = off;
    }
    __syncthreads();

    // --- phase 2: gather 16px x 256ch, sum over z, convert bf16 into LDS
    {
        int c = tid;
        for (int m = 0; m < 16; ++m) {
            float s = 0.f;
            for (int iz = 0; iz < 4; ++iz) {
                int cam = s_cam[m][iz];
                if (cam >= 0)
                    s += fuse[((size_t)(b * NCAM + cam) * CFEAT + c) * NPIX + s_off[m][iz]];
            }
            s_a[m * SA_STRIDE + c] = f2bf(s);
        }
    }
    if (wid == 0) __builtin_amdgcn_s_wait_tensorcnt(0);
    __syncthreads();

    // --- phase 3: WMMA per wave; waves 0..4 cover 80 out channels, K=256
    if (wid < 5) {
        int ncol   = wid * 16 + (lane & 15);
        int khalfA = (lane < 16) ? 0 : 8;
        int kbaseB = (lane < 16) ? 0 : 16;
        int mrowA  = lane & 15;
        const __bf16* arow = &s_a[mrowA * SA_STRIDE];
        const __bf16* brow = &s_w[ncol * SW_STRIDE];
        v8f acc = {};
        for (int K0 = 0; K0 < 256; K0 += 32) {
            v16bf a  = ld_frag(arow + K0 + khalfA, arow + K0 + khalfA + 16);
            v16bf bm = ld_frag(brow + K0 + khalfA, brow + K0 + khalfA + 16);
            acc = __builtin_amdgcn_wmma_f32_16x16x32_bf16(false, a, false, bm,
                                                          (short)0, acc, false, false);
        }
        int o = ncol;
        float inv   = rsqrtf(fnv[o] + EPSBN);
        float scale = fng[o] * inv;
        float shift = fnb[o] + (bfin[o] - fnm[o]) * scale;
        int mbase = (lane < 16) ? 0 : 8;
#pragma unroll
        for (int v = 0; v < 8; ++v)
            s_x[(mbase + v) * 80 + o] = fmaxf(acc[v] * scale + shift, 0.f);
    }
    __syncthreads();

    // --- phase 4: attention conv 80->1 + sigmoid per pixel
    if (tid < 16) {
        float s = attb[0];
        for (int o = 0; o < 80; ++o) s += attw[o] * s_x[tid * 80 + o];
        s_att[tid] = 1.f / (1.f + __expf(-s));
    }
    __syncthreads();

    // --- phase 5: out = att * x, layout (B,80,200,200)
    for (int i = tid; i < 16 * 80; i += 256) {
        int m = i / 80, o = i % 80;
        int p = tile * 16 + m;
        out[((size_t)(b * 80 + o)) * 40000 + p] = s_att[m] * s_x[m * 80 + o];
    }
}

// ---------------- host launch ----------------
extern "C" void kernel_launch(void* const* d_in, const int* in_sizes, int n_in,
                              void* d_out, int out_size, void* d_ws, size_t ws_size,
                              hipStream_t stream) {
    (void)in_sizes; (void)n_in; (void)out_size; (void)ws_size;
    const float* feats     = (const float*)d_in[0];
    const float* pts       = (const float*)d_in[1];
    const float* l2i       = (const float*)d_in[2];
    const float* img_aug   = (const float*)d_in[3];
    const float* lidar_aug = (const float*)d_in[4];
    const float* dw1 = (const float*)d_in[5];  const float* db1 = (const float*)d_in[6];
    const float* dg1 = (const float*)d_in[7];  const float* dbe1= (const float*)d_in[8];
    const float* dm1 = (const float*)d_in[9];  const float* dv1 = (const float*)d_in[10];
    const float* dw2 = (const float*)d_in[11]; const float* db2 = (const float*)d_in[12];
    const float* dg2 = (const float*)d_in[13]; const float* dbe2= (const float*)d_in[14];
    const float* dm2 = (const float*)d_in[15]; const float* dv2 = (const float*)d_in[16];
    const float* fw  = (const float*)d_in[17]; const float* fb  = (const float*)d_in[18];
    const float* fg  = (const float*)d_in[19]; const float* fbe = (const float*)d_in[20];
    const float* fm  = (const float*)d_in[21]; const float* fv  = (const float*)d_in[22];
    const float* finw= (const float*)d_in[23]; const float* finb= (const float*)d_in[24];
    const float* fng = (const float*)d_in[25]; const float* fnb = (const float*)d_in[26];
    const float* fnm = (const float*)d_in[27]; const float* fnv = (const float*)d_in[28];
    const float* attw= (const float*)d_in[29]; const float* attb= (const float*)d_in[30];

    // scratch layout (bytes)
    const size_t SZ_DEPTH = (size_t)NIMG * IMG_H * IMG_W * 4;        //  8,650,752
    const size_t SZ_D1    = (size_t)NIMG * 32 * C1H * C1W * 4;       // 17,301,504
    const size_t SZ_D2    = (size_t)NIMG * 64 * NPIX * 4;            //  8,650,752
    const size_t SZ_FUSE  = (size_t)NIMG * CFEAT * NPIX * 4;         // 34,603,008
    const size_t SZ_PROJ  = (size_t)NIMG * 16 * 4;                   //        768
    const size_t SZ_WBF   = (size_t)80 * 256 * 2;                    //     40,960
    const size_t SZ_WT    = (size_t)9 * 256 * KFUSE * 2;             //  1,474,560

    char* wsb   = (char*)d_ws;
    float*  depth = (float*)(wsb);
    float*  d1    = (float*)(wsb + SZ_DEPTH);
    float*  d2    = (float*)(wsb + SZ_DEPTH + SZ_D1);
    float*  fusep = (float*)(wsb + SZ_DEPTH + SZ_D1 + SZ_D2);
    float*  projp = (float*)(wsb + SZ_DEPTH + SZ_D1 + SZ_D2 + SZ_FUSE);
    __bf16* wbf   = (__bf16*)(wsb + SZ_DEPTH + SZ_D1 + SZ_D2 + SZ_FUSE + SZ_PROJ);
    __bf16* wT    = (__bf16*)(wsb + SZ_DEPTH + SZ_D1 + SZ_D2 + SZ_FUSE + SZ_PROJ + SZ_WBF);
    __bf16* act   = (__bf16*)(wsb + SZ_DEPTH + SZ_D1 + SZ_D2 + SZ_FUSE + SZ_PROJ + SZ_WBF + SZ_WT);

    (void)hipMemsetAsync(depth, 0, SZ_DEPTH, stream);
    precompute_proj<<<1, 32, 0, stream>>>(l2i, img_aug, lidar_aug, projp);

    {
        int total = BATCH * NPTS * NCAM;
        scatter_depth<<<(total + 255) / 256, 256, 0, stream>>>(pts, projp, depth);
    }
    {
        int total = NIMG * 32 * C1H * C1W;
        conv1_kernel<<<(total + 255) / 256, 256, 0, stream>>>(depth, dw1, db1, dg1, dbe1, dm1, dv1, d1);
    }
    {
        int total = NIMG * 64 * NPIX;
        conv2_kernel<<<(total + 255) / 256, 256, 0, stream>>>(d1, dw2, db2, dg2, dbe2, dm2, dv2, d2);
    }
    {
        int total = NIMG * NPIX * KFUSE;
        build_act<<<(total + 255) / 256, 256, 0, stream>>>(d2, feats, act);
    }
    {
        int total = 9 * 256 * KFUSE + 80 * 256;
        convert_weights<<<(total + 255) / 256, 256, 0, stream>>>(fw, finw, wT, wbf);
    }
    fuse_conv_wmma<<<dim3(176, 16, NIMG), 32, 0, stream>>>(act, wT, fb, fg, fbe, fm, fv, fusep);
    bev_final_kernel<<<BATCH * 2500, 256, 0, stream>>>(fusep, projp, wbf, finb, fng, fnb, fnm, fnv,
                                                       attw, attb, (float*)d_out);
}
